// TransformerDecoder_9131100472001
// MI455X (gfx1250) — compile-verified, matrix operand order checked by reading, not run on hardware
//
#include <hip/hip_runtime.h>
#include <hip/hip_bf16.h>

#define NB 2
#define NS 2048
#define ND 1024
#define NH 16
#define NL 4
#define NV 32000
#define NDFF 4096
#define NDH 64
#define NM (NB * NS)   // 4096 token rows

typedef __attribute__((ext_vector_type(16))) __bf16 v16bf;
typedef __attribute__((ext_vector_type(8)))  float  v8f;

// ---------------------------------------------------------------------------
// helpers
// ---------------------------------------------------------------------------
__device__ __forceinline__ unsigned short f2bf(float x) {
  unsigned u = __float_as_uint(x);
  u += 0x7FFFu + ((u >> 16) & 1u);      // round-to-nearest-even
  return (unsigned short)(u >> 16);
}
__device__ __forceinline__ unsigned pack2(float a, float b) {
  return (unsigned)f2bf(a) | ((unsigned)f2bf(b) << 16);
}

// Load one 16x32 bf16 WMMA operand fragment from an LDS tile stored row-major
// (rows = M for the A operand; rows = N for the B operand, i.e. tile holds B^T).
// CDNA5 16-bit A layout: lanes 0-15 row=lane, K 0..7 in v0..3, K 16..23 in v4..7;
// lanes 16-31 row=lane-16, K 8..15 in v0..3, K 24..31 in v4..7.
__device__ __forceinline__ v16bf frag_ld(const unsigned short* tile, int row0,
                                         int k0, int stride) {
  const int lane = threadIdx.x & 31;
  const unsigned* p =
      (const unsigned*)(tile + (row0 + (lane & 15)) * stride);
  const int kh = (lane >> 4) << 3;      // 0 or 8
  union { v16bf v; unsigned u[8]; } f;
#pragma unroll
  for (int i = 0; i < 4; ++i) f.u[i]     = p[(k0 + kh +       2 * i) >> 1];
#pragma unroll
  for (int i = 0; i < 4; ++i) f.u[4 + i] = p[(k0 + kh + 16 + 2 * i) >> 1];
  return f.v;
}

// ---------------------------------------------------------------------------
// GEMM: C[M,N] = act( A[M,K] * B[K,N] + bias[N] (+ res[M,N]) )
// fp32 in/out, bf16 WMMA math. Block tile 128x128, K-step 64, 256 threads.
// ---------------------------------------------------------------------------
#define GLDS 72   // padded LDS row stride (elements)

__global__ __launch_bounds__(256) void gemm_bf16_wmma(
    const float* __restrict__ A, const float* __restrict__ Bm,
    const float* __restrict__ bias, const float* __restrict__ res,
    float* __restrict__ C, int M, int N, int K, int relu) {
  __shared__ unsigned short As[128 * GLDS];   // [m][k]
  __shared__ unsigned short Bs[128 * GLDS];   // B^T tile: [n][k]
  const int tid  = threadIdx.x;
  const int lane = tid & 31;
  const int wave = tid >> 5;
  const int wm   = wave >> 2;   // 0..1 : 64 rows each
  const int wn   = wave & 3;    // 0..3 : 32 cols each
  const int bm   = blockIdx.y * 128;
  const int bn   = blockIdx.x * 128;

  v8f acc[4][2];
  {
    v8f z = {};
#pragma unroll
    for (int i = 0; i < 4; ++i)
#pragma unroll
      for (int j = 0; j < 2; ++j) acc[i][j] = z;
  }

  for (int k0 = 0; k0 < K; k0 += 64) {
    // stage A tile: 128x64 fp32 -> bf16, coalesced float4 reads
#pragma unroll
    for (int p2 = 0; p2 < 8; ++p2) {
      int idx = p2 * 1024 + tid * 4;
      int r = idx >> 6, c = idx & 63;
      const float4 f = *(const float4*)(A + (size_t)(bm + r) * K + k0 + c);
      unsigned* d = (unsigned*)(As + r * GLDS + c);
      d[0] = pack2(f.x, f.y);
      d[1] = pack2(f.z, f.w);
    }
    // stage B tile transposed: 64x128 fp32 -> bf16 [n][k]
#pragma unroll
    for (int p2 = 0; p2 < 8; ++p2) {
      int idx = p2 * 1024 + tid * 4;
      int kr = idx >> 7, c = idx & 127;
      const float4 f = *(const float4*)(Bm + (size_t)(k0 + kr) * N + bn + c);
      Bs[(c + 0) * GLDS + kr] = f2bf(f.x);
      Bs[(c + 1) * GLDS + kr] = f2bf(f.y);
      Bs[(c + 2) * GLDS + kr] = f2bf(f.z);
      Bs[(c + 3) * GLDS + kr] = f2bf(f.w);
    }
    __syncthreads();
#pragma unroll
    for (int kk = 0; kk < 64; kk += 32) {
      v16bf af[4], bfr[2];
#pragma unroll
      for (int mt = 0; mt < 4; ++mt)
        af[mt] = frag_ld(As, wm * 64 + mt * 16, kk, GLDS);
#pragma unroll
      for (int nt = 0; nt < 2; ++nt)
        bfr[nt] = frag_ld(Bs, wn * 32 + nt * 16, kk, GLDS);
#pragma unroll
      for (int mt = 0; mt < 4; ++mt)
#pragma unroll
        for (int nt = 0; nt < 2; ++nt)
          acc[mt][nt] = __builtin_amdgcn_wmma_f32_16x16x32_bf16(
              false, af[mt], false, bfr[nt], (short)0, acc[mt][nt],
              false, false);
    }
    __syncthreads();
  }

  // epilogue: C-layout rows j (+8 for upper lane half), cols = lane&15
#pragma unroll
  for (int mt = 0; mt < 4; ++mt)
#pragma unroll
    for (int nt = 0; nt < 2; ++nt) {
      int gc = bn + wn * 32 + nt * 16 + (lane & 15);
      float bv = bias ? bias[gc] : 0.0f;
      int rb = bm + wm * 64 + mt * 16 + ((lane >> 4) << 3);
#pragma unroll
      for (int j = 0; j < 8; ++j) {
        size_t o = (size_t)(rb + j) * N + gc;
        float vv = acc[mt][nt][j] + bv;
        if (res) vv += res[o];
        if (relu) vv = fmaxf(vv, 0.0f);
        C[o] = vv;
      }
    }
}

// ---------------------------------------------------------------------------
// Flash attention: one block per (b, h, 64-query tile), 4 waves.
// Wave w owns query rows [w*16, w*16+16). Causal; key loop stops at diagonal.
// ---------------------------------------------------------------------------
__global__ __launch_bounds__(128) void attn_kernel(
    const float* __restrict__ q, const float* __restrict__ k,
    const float* __restrict__ v, float* __restrict__ o) {
  __shared__ unsigned short Qs[64 * 72];   // [q][d], pre-scaled by 1/8
  __shared__ unsigned short Ks[64 * 72];   // [key][d]  (== B^T for Q*K^T)
  __shared__ unsigned short Vt[64 * 72];   // [d][key]  (== B^T for P*V)
  __shared__ unsigned short Ps[4][16 * 72];// per-wave P strip (A for P*V)
  const int tid  = threadIdx.x;
  const int lane = tid & 31;
  const int w    = tid >> 5;
  const int q0   = blockIdx.x * 64;
  const int h    = blockIdx.y;
  const int b    = blockIdx.z;
  const size_t hd = (size_t)h * NDH;

#pragma unroll
  for (int p = 0; p < 32; ++p) {
    int idx = p * 128 + tid;
    int r = idx >> 6, c = idx & 63;
    float qv = q[((size_t)b * NS + q0 + r) * ND + hd + c] * 0.125f; // 1/sqrt(64)
    Qs[r * 72 + c] = f2bf(qv);
  }

  float mrow[8], lrow[8];
#pragma unroll
  for (int j = 0; j < 8; ++j) { mrow[j] = -3.0e38f; lrow[j] = 0.0f; }
  v8f oacc[4];
  {
    v8f z = {};
#pragma unroll
    for (int dt = 0; dt < 4; ++dt) oacc[dt] = z;
  }
  const int rloc = (lane >> 4) << 3;   // row base (+j) within the 16-row strip

  for (int kt = 0; kt <= q0; kt += 64) {
    __syncthreads();
#pragma unroll
    for (int p = 0; p < 32; ++p) {
      int idx = p * 128 + tid;
      int r = idx >> 6, c = idx & 63;
      size_t gi = ((size_t)b * NS + kt + r) * ND + hd + c;
      Ks[r * 72 + c] = f2bf(k[gi]);
      Vt[c * 72 + r] = f2bf(v[gi]);
    }
    __syncthreads();

    // S = Q * K^T  (16x64 per wave)
    v8f sac[4];
    {
      v8f z = {};
#pragma unroll
      for (int nt = 0; nt < 4; ++nt) sac[nt] = z;
    }
#pragma unroll
    for (int kk = 0; kk < 64; kk += 32) {
      v16bf aq = frag_ld(Qs, w * 16, kk, 72);
#pragma unroll
      for (int nt = 0; nt < 4; ++nt) {
        v16bf bk = frag_ld(Ks, nt * 16, kk, 72);
        sac[nt] = __builtin_amdgcn_wmma_f32_16x16x32_bf16(
            false, aq, false, bk, (short)0, sac[nt], false, false);
      }
    }
    if (kt == q0) {   // causal mask only needed on the diagonal tile
      int qg = q0 + w * 16 + rloc;
#pragma unroll
      for (int nt = 0; nt < 4; ++nt) {
        int kg = kt + nt * 16 + (lane & 15);
#pragma unroll
        for (int j = 0; j < 8; ++j)
          if (kg > qg + j) sac[nt][j] = -1.0e9f;
      }
    }
    // online softmax; rows 0-7 live in lanes 0-15, rows 8-15 in lanes 16-31
#pragma unroll
    for (int j = 0; j < 8; ++j) {
      float mx = fmaxf(fmaxf(sac[0][j], sac[1][j]), fmaxf(sac[2][j], sac[3][j]));
#pragma unroll
      for (int off = 8; off >= 1; off >>= 1) mx = fmaxf(mx, __shfl_xor(mx, off));
      float mnew  = fmaxf(mrow[j], mx);
      float alpha = __expf(mrow[j] - mnew);
      float ssum  = 0.0f;
#pragma unroll
      for (int nt = 0; nt < 4; ++nt) {
        float pv = __expf(sac[nt][j] - mnew);
        sac[nt][j] = pv;
        ssum += pv;
      }
#pragma unroll
      for (int off = 8; off >= 1; off >>= 1) ssum += __shfl_xor(ssum, off);
      lrow[j] = lrow[j] * alpha + ssum;
      mrow[j] = mnew;
#pragma unroll
      for (int dt = 0; dt < 4; ++dt) oacc[dt][j] *= alpha;
    }
    // spill P to the per-wave LDS strip, re-fragment as A operand
#pragma unroll
    for (int nt = 0; nt < 4; ++nt)
#pragma unroll
      for (int j = 0; j < 8; ++j)
        Ps[w][(rloc + j) * 72 + nt * 16 + (lane & 15)] = f2bf(sac[nt][j]);
    // O += P * V
#pragma unroll
    for (int kk = 0; kk < 64; kk += 32) {
      v16bf ap = frag_ld(Ps[w], 0, kk, 72);
#pragma unroll
      for (int dt = 0; dt < 4; ++dt) {
        v16bf bv = frag_ld(Vt, dt * 16, kk, 72);
        oacc[dt] = __builtin_amdgcn_wmma_f32_16x16x32_bf16(
            false, ap, false, bv, (short)0, oacc[dt], false, false);
      }
    }
  }
  // write O / l
#pragma unroll
  for (int dt = 0; dt < 4; ++dt)
#pragma unroll
    for (int j = 0; j < 8; ++j) {
      int qr = q0 + w * 16 + rloc + j;
      o[((size_t)b * NS + qr) * ND + hd + dt * 16 + (lane & 15)] =
          oacc[dt][j] / lrow[j];
    }
}

// ---------------------------------------------------------------------------
// Embedding + sinusoidal PE: x = tok_emb[id]*sqrt(D) + pe(s, d)
// ---------------------------------------------------------------------------
__global__ __launch_bounds__(256) void embed_kernel(
    const int* __restrict__ ids, const float* __restrict__ emb,
    float* __restrict__ x) {
  const int t  = blockIdx.x;        // token index in [0, B*S)
  const int s  = t & (NS - 1);
  const int id = ids[t];
#pragma unroll
  for (int i = 0; i < 4; ++i) {
    int d = threadIdx.x + i * 256;
    float val = emb[(size_t)id * ND + d] * 32.0f;   // sqrt(1024)
    float e  = (float)(d & ~1) * (-9.210340371976184f / (float)ND); // -ln(1e4)/D
    float ph = (float)s * __expf(e);
    val += (d & 1) ? cosf(ph) : sinf(ph);
    x[(size_t)t * ND + d] = val;
  }
}

// ---------------------------------------------------------------------------
// LayerNorm over D=1024, one block per row
// ---------------------------------------------------------------------------
__global__ __launch_bounds__(256) void ln_kernel(
    const float* __restrict__ in, const float* __restrict__ g,
    const float* __restrict__ bta, float* __restrict__ out) {
  __shared__ float s1[8], s2[8];
  const int row = blockIdx.x;
  const float* r = in + (size_t)row * ND;
  float vv[4];
  float s = 0.0f;
#pragma unroll
  for (int i = 0; i < 4; ++i) { vv[i] = r[threadIdx.x + i * 256]; s += vv[i]; }
#pragma unroll
  for (int off = 16; off >= 1; off >>= 1) s += __shfl_xor(s, off);
  if ((threadIdx.x & 31) == 0) s1[threadIdx.x >> 5] = s;
  __syncthreads();
  float tot = 0.0f;
#pragma unroll
  for (int wv = 0; wv < 8; ++wv) tot += s1[wv];
  float mu = tot * (1.0f / ND);
  float vs = 0.0f;
#pragma unroll
  for (int i = 0; i < 4; ++i) { float d = vv[i] - mu; vs += d * d; }
#pragma unroll
  for (int off = 16; off >= 1; off >>= 1) vs += __shfl_xor(vs, off);
  if ((threadIdx.x & 31) == 0) s2[threadIdx.x >> 5] = vs;
  __syncthreads();
  float vtot = 0.0f;
#pragma unroll
  for (int wv = 0; wv < 8; ++wv) vtot += s2[wv];
  float rstd = rsqrtf(vtot * (1.0f / ND) + 1e-5f);
#pragma unroll
  for (int i = 0; i < 4; ++i) {
    int d = threadIdx.x + i * 256;
    out[(size_t)row * ND + d] = (vv[i] - mu) * rstd * g[d] + bta[d];
  }
}

// ---------------------------------------------------------------------------
// host orchestration
// ---------------------------------------------------------------------------
extern "C" void kernel_launch(void* const* d_in, const int* in_sizes, int n_in,
                              void* d_out, int out_size, void* d_ws,
                              size_t ws_size, hipStream_t stream) {
  const int*   ids  = (const int*)  d_in[0];
  const float* emb  = (const float*)d_in[1];
  const float* Wq   = (const float*)d_in[2];
  const float* bq   = (const float*)d_in[3];
  const float* Wk   = (const float*)d_in[4];
  const float* bk   = (const float*)d_in[5];
  const float* Wv   = (const float*)d_in[6];
  const float* bv   = (const float*)d_in[7];
  const float* Wo   = (const float*)d_in[8];
  const float* bo   = (const float*)d_in[9];
  const float* g1   = (const float*)d_in[10];
  const float* b1n  = (const float*)d_in[11];
  const float* W1   = (const float*)d_in[12];
  const float* b1f  = (const float*)d_in[13];
  const float* W2   = (const float*)d_in[14];
  const float* b2f  = (const float*)d_in[15];
  const float* g3   = (const float*)d_in[16];
  const float* b3n  = (const float*)d_in[17];
  const float* Wout = (const float*)d_in[18];
  const float* bout = (const float*)d_in[19];
  float* out = (float*)d_out;

  const size_t MD = (size_t)NM * ND;
  float* x  = (float*)d_ws;   // [4096,1024]
  float* qb = x  + MD;
  float* kb = qb + MD;
  float* vb = kb + MD;
  float* ab = vb + MD;
  float* tb = ab + MD;
  float* f1 = tb + MD;        // [4096,4096]

  embed_kernel<<<NM, 256, 0, stream>>>(ids, emb, x);

  dim3 gP  (ND   / 128, NM / 128);
  dim3 gF1 (NDFF / 128, NM / 128);
  dim3 gOut(NV   / 128, NM / 128);
  dim3 gAt (NS / 64, NH, NB);

  for (int l = 0; l < NL; ++l) {
    const float* wq = Wq + (size_t)l * ND * ND;
    const float* wk = Wk + (size_t)l * ND * ND;
    const float* wv = Wv + (size_t)l * ND * ND;
    const float* wo = Wo + (size_t)l * ND * ND;
    gemm_bf16_wmma<<<gP, 256, 0, stream>>>(x, wq, bq + l * ND, nullptr, qb,
                                           NM, ND, ND, 0);
    gemm_bf16_wmma<<<gP, 256, 0, stream>>>(x, wk, bk + l * ND, nullptr, kb,
                                           NM, ND, ND, 0);
    gemm_bf16_wmma<<<gP, 256, 0, stream>>>(x, wv, bv + l * ND, nullptr, vb,
                                           NM, ND, ND, 0);
    attn_kernel<<<gAt, 128, 0, stream>>>(qb, kb, vb, ab);
    gemm_bf16_wmma<<<gP, 256, 0, stream>>>(ab, wo, bo + l * ND, x, tb,
                                           NM, ND, ND, 0);
    ln_kernel<<<NM, 256, 0, stream>>>(tb, g1 + l * ND, b1n + l * ND, x);
    gemm_bf16_wmma<<<gF1, 256, 0, stream>>>(x, W1 + (size_t)l * ND * NDFF,
                                            b1f + l * NDFF, nullptr, f1,
                                            NM, NDFF, ND, 1);
    gemm_bf16_wmma<<<gP, 256, 0, stream>>>(f1, W2 + (size_t)l * NDFF * ND,
                                           b2f + l * ND, x, tb,
                                           NM, ND, NDFF, 0);
    ln_kernel<<<NM, 256, 0, stream>>>(tb, g3 + l * ND, b3n + l * ND, x);
  }
  gemm_bf16_wmma<<<gOut, 256, 0, stream>>>(x, Wout, bout, nullptr, out,
                                           NM, NV, ND, 0);
  (void)in_sizes; (void)n_in; (void)out_size; (void)ws_size;
}
